// SelfAttention_31568009625956
// MI455X (gfx1250) — matrix operand order repaired and checked by reading, not compiled, not run on hardware
//
#include <hip/hip_runtime.h>

// ---------------------------------------------------------------------------
// Types for CDNA5 WMMA (wave32, 16x16x32 bf16 -> f32)
// ---------------------------------------------------------------------------
typedef __attribute__((ext_vector_type(16))) __bf16 v16bf;
typedef __attribute__((ext_vector_type(8)))  float  v8f;

union BF16x16 { unsigned int u[8]; v16bf v; };

__device__ __forceinline__ unsigned short f2bf(float f) {
    unsigned int u = __builtin_bit_cast(unsigned int, f);
    u += 0x7FFFu + ((u >> 16) & 1u);          // round-to-nearest-even
    return (unsigned short)(u >> 16);
}
__device__ __forceinline__ unsigned int pack2(float a, float b) {
    return (unsigned int)f2bf(a) | ((unsigned int)f2bf(b) << 16);
}
__device__ __forceinline__ v16bf mk_frag(uint4 lo, uint4 hi) {
    BF16x16 t;
    t.u[0] = lo.x; t.u[1] = lo.y; t.u[2] = lo.z; t.u[3] = lo.w;
    t.u[4] = hi.x; t.u[5] = hi.y; t.u[6] = hi.z; t.u[7] = hi.w;
    return t.v;
}
__device__ __forceinline__ v8f wmma_bf16(v16bf a, v16bf b, v8f c) {
    return __builtin_amdgcn_wmma_f32_16x16x32_bf16(false, a, false, b,
                                                   (short)0, c, false, false);
}

#define SEQ   2048
#define SCALE 0.125f   // 64^-0.5

// ---------------------------------------------------------------------------
// Prep kernels
// ---------------------------------------------------------------------------
__global__ void cvt_bf16_kernel(const float* __restrict__ in,
                                unsigned short* __restrict__ out, int n) {
    int i = blockIdx.x * 256 + threadIdx.x;
    if (i < n) out[i] = f2bf(in[i]);
}

// in: [K][N] f32 row-major  ->  out: [N][K] bf16 row-major
__global__ void transpose_bf16_kernel(const float* __restrict__ in,
                                      unsigned short* __restrict__ out,
                                      int K, int N) {
    int i = blockIdx.x * 256 + threadIdx.x;
    if (i < K * N) {
        int k = i / N, n = i - k * N;
        out[(size_t)n * K + k] = f2bf(in[i]);
    }
}

// Pack int32 0/1 mask into bitmask words: word i holds mask[i*32 .. i*32+31]
__global__ void pack_mask_kernel(const int* __restrict__ in,
                                 unsigned int* __restrict__ out, int nwords) {
    int i = blockIdx.x * 256 + threadIdx.x;
    if (i < nwords) {
        const int* p = in + (size_t)i * 32;
        unsigned int w = 0;
#pragma unroll
        for (int j = 0; j < 32; ++j) w |= (p[j] != 0 ? 1u : 0u) << j;
        out[i] = w;
    }
}

// ---------------------------------------------------------------------------
// Stage 1: qkv = x @ W_qkv   (A bf16 [4096][1024], Bt bf16 [3072][1024])
// One wave computes a 32(M) x 64(N) block: two A-fragments share each
// B-fragment. Scatter: q,k -> [b,h,L,64] ; v -> transposed [b,h,64,L].
// ---------------------------------------------------------------------------
__global__ __launch_bounds__(256) void gemm_qkv_kernel(
    const unsigned short* __restrict__ A,
    const unsigned short* __restrict__ Bt,
    unsigned short* __restrict__ qb,
    unsigned short* __restrict__ kb,
    unsigned short* __restrict__ vt) {
    const int lane = threadIdx.x & 31;
    const int wid  = threadIdx.x >> 5;
    const int l16  = lane & 15;
    const int hi   = (lane >> 4) & 1;
    const int m0   = (blockIdx.y * 8 + wid) * 32;
    const int n0   = blockIdx.x * 64;
    const int K    = 1024;

    const v8f vz = {0.f,0.f,0.f,0.f,0.f,0.f,0.f,0.f};
    v8f acc[2][4] = {{vz, vz, vz, vz}, {vz, vz, vz, vz}};

    for (int k0 = 0; k0 < K; k0 += 32) {
        const uint4* ap0 = (const uint4*)(A + (size_t)(m0 + l16) * K + k0 + hi * 8);
        const uint4* ap1 = (const uint4*)(A + (size_t)(m0 + 16 + l16) * K + k0 + hi * 8);
        v16bf af0 = mk_frag(ap0[0], ap0[2]);
        v16bf af1 = mk_frag(ap1[0], ap1[2]);
#pragma unroll
        for (int t = 0; t < 4; ++t) {
            const uint4* bp = (const uint4*)(Bt + (size_t)(n0 + 16 * t + l16) * K + k0 + hi * 8);
            v16bf bf = mk_frag(bp[0], bp[2]);
            acc[0][t] = wmma_bf16(af0, bf, acc[0][t]);
            acc[1][t] = wmma_bf16(af1, bf, acc[1][t]);
        }
    }
#pragma unroll
    for (int t = 0; t < 4; ++t) {
        int col   = n0 + 16 * t + l16;
        int which = col >> 10;
        int hh    = (col >> 6) & 15;
        int dc    = col & 63;
#pragma unroll
        for (int u = 0; u < 2; ++u) {
#pragma unroll
            for (int r = 0; r < 8; ++r) {
                int row = m0 + 16 * u + r + hi * 8;   // global token
                int bb  = row >> 11;
                int l   = row & 2047;
                unsigned short v = f2bf(acc[u][t][r]);
                if (which == 0)
                    qb[((size_t)(bb * 16 + hh) * 2048 + l) * 64 + dc] = v;
                else if (which == 1)
                    kb[((size_t)(bb * 16 + hh) * 2048 + l) * 64 + dc] = v;
                else
                    vt[((size_t)(bb * 16 + hh) * 64 + dc) * 2048 + l] = v;
            }
        }
    }
}

// ---------------------------------------------------------------------------
// Stage 2: flash-style attention, one wave per (b,h, 32-query tile).
// Computes S^T = K_tile @ Q^T so that the C-layout of S^T equals the
// A-layout required for P @ V -- probabilities never leave registers.
// K/V fragments are shared by both 16-query subtiles.
// ---------------------------------------------------------------------------
__global__ __launch_bounds__(256) void attn_kernel(
    const unsigned short* __restrict__ qb,
    const unsigned short* __restrict__ kb,
    const unsigned short* __restrict__ vt,
    const unsigned int* __restrict__ mpack,
    unsigned short* __restrict__ zb) {
    const int lane = threadIdx.x & 31;
    const int l16  = lane & 15;
    const int hi   = (lane >> 4) & 1;
    const int wgid = blockIdx.x * 8 + (threadIdx.x >> 5);  // 0..2047
    const int bh   = wgid >> 6;     // 0..31
    const int qt   = wgid & 63;
    const int b    = bh >> 4;
    const int h    = bh & 15;
    const int q0   = qt << 5;       // 32 queries per wave

    // Q^T B-fragments, two 16-query subtiles, reused across all keys
    v16bf bq[2][2];
#pragma unroll
    for (int u = 0; u < 2; ++u) {
        const unsigned short* qrow = qb + (size_t)(bh * 2048 + q0 + 16 * u + l16) * 64;
        const uint4* qp0 = (const uint4*)(qrow + hi * 8);
        const uint4* qp1 = (const uint4*)(qrow + 32 + hi * 8);
        bq[u][0] = mk_frag(qp0[0], qp0[2]);   // feature dims 0..31
        bq[u][1] = mk_frag(qp1[0], qp1[2]);   // feature dims 32..63
    }

    const v8f vz = {0.f,0.f,0.f,0.f,0.f,0.f,0.f,0.f};
    v8f acc[2][4] = {{vz, vz, vz, vz}, {vz, vz, vz, vz}};
    float mx[2]   = {-1e30f, -1e30f};
    float lall[2] = {0.f, 0.f};
    float lmsk[2] = {0.f, 0.f};

    const unsigned int* mw0 = mpack + (size_t)(b * 2048 + q0 + l16) * 64;      // 64 words/row
    const unsigned int* mw1 = mw0 + 16 * 64;

    for (int j0 = 0; j0 < SEQ; j0 += 32) {
        // --- K A-fragments for keys j0..j0+15 and j0+16..j0+31 (shared) ---
        const unsigned short* kr0 = kb + (size_t)(bh * 2048 + j0 + l16) * 64;
        const unsigned short* kr1 = kr0 + 16 * 64;
        v16bf ak00 = mk_frag(*(const uint4*)(kr0 + hi * 8),      *(const uint4*)(kr0 + hi * 8 + 16));
        v16bf ak01 = mk_frag(*(const uint4*)(kr0 + 32 + hi * 8), *(const uint4*)(kr0 + 32 + hi * 8 + 16));
        v16bf ak10 = mk_frag(*(const uint4*)(kr1 + hi * 8),      *(const uint4*)(kr1 + hi * 8 + 16));
        v16bf ak11 = mk_frag(*(const uint4*)(kr1 + 32 + hi * 8), *(const uint4*)(kr1 + 32 + hi * 8 + 16));

        v16bf apfrag[2];
#pragma unroll
        for (int u = 0; u < 2; ++u) {
            v8f c0 = wmma_bf16(ak00, bq[u][0], wmma_bf16(ak01, bq[u][1], vz));
            v8f c1 = wmma_bf16(ak10, bq[u][0], wmma_bf16(ak11, bq[u][1], vz));

            // --- streaming softmax update (row == lane's query) ---
            float s0[8], s1[8];
            float mloc = -1e30f;
#pragma unroll
            for (int r = 0; r < 8; ++r) {
                s0[r] = c0[r] * SCALE;
                s1[r] = c1[r] * SCALE;
                mloc = fmaxf(mloc, fmaxf(s0[r], s1[r]));
            }
            mloc = fmaxf(mloc, __shfl_xor(mloc, 16, 32));
            float nm = fmaxf(mx[u], mloc);
            float fc = __expf(mx[u] - nm);
            mx[u] = nm;

            // mask bits for this lane: keys j0 + hi*8 + {0..7} and +16
            unsigned int wbits = (u == 0 ? mw0 : mw1)[j0 >> 5];

            float sa = 0.f, sm = 0.f;
            unsigned int pa[8];
#pragma unroll
            for (int i = 0; i < 4; ++i) {
                float p0 = __expf(s0[2 * i] - nm), p1 = __expf(s0[2 * i + 1] - nm);
                sa += p0 + p1;
                float w0 = ((wbits >> (hi * 8 + 2 * i)) & 1u) ? p0 : 0.f;
                float w1 = ((wbits >> (hi * 8 + 2 * i + 1)) & 1u) ? p1 : 0.f;
                sm += w0 + w1;
                pa[i] = pack2(w0, w1);
                float p2 = __expf(s1[2 * i] - nm), p3 = __expf(s1[2 * i + 1] - nm);
                sa += p2 + p3;
                float w2 = ((wbits >> (16 + hi * 8 + 2 * i)) & 1u) ? p2 : 0.f;
                float w3 = ((wbits >> (16 + hi * 8 + 2 * i + 1)) & 1u) ? p3 : 0.f;
                sm += w2 + w3;
                pa[4 + i] = pack2(w2, w3);
            }
            sa += __shfl_xor(sa, 16, 32);
            sm += __shfl_xor(sm, 16, 32);
            lall[u] = lall[u] * fc + sa;
            lmsk[u] = lmsk[u] * fc + sm;

            BF16x16 pf;
#pragma unroll
            for (int i = 0; i < 8; ++i) pf.u[i] = pa[i];
            apfrag[u] = pf.v;   // A-fragment of P: 16 queries x 32 keys

            // rescale running z accumulator (row r+8*hi, stats live in lane q)
#pragma unroll
            for (int r = 0; r < 8; ++r) {
                float fb = __shfl(fc, r + hi * 8, 32);
                acc[u][0][r] *= fb; acc[u][1][r] *= fb;
                acc[u][2][r] *= fb; acc[u][3][r] *= fb;
            }
        }

        // --- z += P @ V (V stored transposed [bh][d][L]; fragments shared) ---
#pragma unroll
        for (int t = 0; t < 4; ++t) {
            const unsigned short* vrow =
                vt + (size_t)(bh * 64 + 16 * t + l16) * 2048 + j0 + hi * 8;
            v16bf bv = mk_frag(*(const uint4*)vrow, *(const uint4*)(vrow + 16));
            acc[0][t] = wmma_bf16(apfrag[0], bv, acc[0][t]);
            acc[1][t] = wmma_bf16(apfrag[1], bv, acc[1][t]);
        }
    }

    // attn = m*e / (Z_masked + 1e-10 * Z_all)   (matches reference algebra)
#pragma unroll
    for (int u = 0; u < 2; ++u) {
        float inv = 1.0f / (lmsk[u] + 1e-10f * lall[u]);
#pragma unroll
        for (int r = 0; r < 8; ++r) {
            float ib = __shfl(inv, r + hi * 8, 32);
            int tok = b * 2048 + q0 + 16 * u + r + hi * 8;
#pragma unroll
            for (int t = 0; t < 4; ++t) {
                zb[(size_t)tok * 1024 + h * 64 + 16 * t + l16] = f2bf(acc[u][t][r] * ib);
            }
        }
    }
}

// ---------------------------------------------------------------------------
// Stage 3: out = z @ W_out + b_out   (A bf16 [4096][1024], Bt bf16 [1024][1024])
// 32(M) x 64(N) per wave.
// ---------------------------------------------------------------------------
__global__ __launch_bounds__(256) void gemm_out_kernel(
    const unsigned short* __restrict__ A,
    const unsigned short* __restrict__ Bt,
    const float* __restrict__ bias,
    float* __restrict__ out) {
    const int lane = threadIdx.x & 31;
    const int wid  = threadIdx.x >> 5;
    const int l16  = lane & 15;
    const int hi   = (lane >> 4) & 1;
    const int m0   = (blockIdx.y * 8 + wid) * 32;
    const int n0   = blockIdx.x * 64;
    const int K    = 1024;

    const v8f vz = {0.f,0.f,0.f,0.f,0.f,0.f,0.f,0.f};
    v8f acc[2][4] = {{vz, vz, vz, vz}, {vz, vz, vz, vz}};

    for (int k0 = 0; k0 < K; k0 += 32) {
        const uint4* ap0 = (const uint4*)(A + (size_t)(m0 + l16) * K + k0 + hi * 8);
        const uint4* ap1 = (const uint4*)(A + (size_t)(m0 + 16 + l16) * K + k0 + hi * 8);
        v16bf af0 = mk_frag(ap0[0], ap0[2]);
        v16bf af1 = mk_frag(ap1[0], ap1[2]);
#pragma unroll
        for (int t = 0; t < 4; ++t) {
            const uint4* bp = (const uint4*)(Bt + (size_t)(n0 + 16 * t + l16) * K + k0 + hi * 8);
            v16bf bf = mk_frag(bp[0], bp[2]);
            acc[0][t] = wmma_bf16(af0, bf, acc[0][t]);
            acc[1][t] = wmma_bf16(af1, bf, acc[1][t]);
        }
    }
#pragma unroll
    for (int t = 0; t < 4; ++t) {
        int col  = n0 + 16 * t + l16;
        float bv = bias[col];
#pragma unroll
        for (int u = 0; u < 2; ++u) {
#pragma unroll
            for (int r = 0; r < 8; ++r) {
                int row = m0 + 16 * u + r + hi * 8;
                out[(size_t)row * 1024 + col] = acc[u][t][r] + bv;
            }
        }
    }
}

// ---------------------------------------------------------------------------
// Host-side launcher
// ---------------------------------------------------------------------------
extern "C" void kernel_launch(void* const* d_in, const int* in_sizes, int n_in,
                              void* d_out, int out_size, void* d_ws, size_t ws_size,
                              hipStream_t stream) {
    const float* x    = (const float*)d_in[0];
    const int*   mask = (const int*)d_in[1];
    const float* Wqkv = (const float*)d_in[2];
    const float* Wout = (const float*)d_in[3];
    const float* bout = (const float*)d_in[4];
    float*       out  = (float*)d_out;

    char* w = (char*)d_ws;
    const size_t MB = 1024 * 1024;
    unsigned short* xb  = (unsigned short*)(w + 0 * MB);   // x bf16        [4096][1024]  8MB
    unsigned short* wqt = (unsigned short*)(w + 8 * MB);   // Wqkv^T bf16   [3072][1024]  6MB
    unsigned short* wot = (unsigned short*)(w + 14 * MB);  // Wout^T bf16   [1024][1024]  2MB
    unsigned short* qb  = (unsigned short*)(w + 16 * MB);  // Q [b,h,L,64]                8MB
    unsigned short* kbv = (unsigned short*)(w + 24 * MB);  // K [b,h,L,64]                8MB
    unsigned short* vtb = (unsigned short*)(w + 32 * MB);  // V^T [b,h,64,L]              8MB
    unsigned short* zb  = (unsigned short*)(w + 40 * MB);  // z [4096][1024]              8MB
    unsigned int*  mpk  = (unsigned int*)(w + 48 * MB);    // packed mask                 1MB

    cvt_bf16_kernel<<<(4096 * 1024) / 256, 256, 0, stream>>>(x, xb, 4096 * 1024);
    transpose_bf16_kernel<<<(1024 * 3072 + 255) / 256, 256, 0, stream>>>(Wqkv, wqt, 1024, 3072);
    transpose_bf16_kernel<<<(1024 * 1024 + 255) / 256, 256, 0, stream>>>(Wout, wot, 1024, 1024);
    pack_mask_kernel<<<(2 * 2048 * 64) / 256, 256, 0, stream>>>(mask, mpk, 2 * 2048 * 64);

    gemm_qkv_kernel<<<dim3(48, 16), 256, 0, stream>>>(xb, wqt, qb, kbv, vtb);
    attn_kernel<<<256, 256, 0, stream>>>(qb, kbv, vtb, mpk, zb);
    gemm_out_kernel<<<dim3(16, 16), 256, 0, stream>>>(zb, wot, bout, out);
}
